// SparseAutoencoder_all_64982855188693
// MI455X (gfx1250) — matrix operand order repaired and checked by reading, not compiled, not run on hardware
//
#include <hip/hip_runtime.h>

// ---------------------------------------------------------------------------
// SparseAutoencoder: encoded = sigmoid(feat @ W1[k]^T + b1[k])
//                    decoded = sigmoid(encoded[:,k] @ W2[k]^T + b2[k])
// B=1024, K=100, F=640, L=128. bf16 WMMA, fp32 accumulate.
// Double-buffered LDS weight slabs (direct shared-array indexing so stores
// stay ds_store_b128, not flat), in-place accumulators, NT stores for the
// 317MB write-once f32 outputs so weights/activations stay L2-resident.
// ---------------------------------------------------------------------------

#define B_DIM 1024
#define K_CLS 100
#define F_DIM 640
#define L_DIM 128
#define LDS_STRIDE 40   // 32 payload + 8 pad (ushorts) to spread LDS banks

typedef __attribute__((ext_vector_type(16))) __bf16        v16bf;
typedef __attribute__((ext_vector_type(8)))  float         v8f;
typedef __attribute__((ext_vector_type(8)))  unsigned short us8;
typedef __attribute__((ext_vector_type(4)))  unsigned short us4;
typedef __attribute__((ext_vector_type(4)))  float          f4;

union FragBF {
    v16bf v;
    us8   u[2];
};

__device__ __forceinline__ unsigned short f32_to_bf16(float f) {
    unsigned int u = __float_as_uint(f);
    u += 0x7FFFu + ((u >> 16) & 1u);   // round-to-nearest-even
    return (unsigned short)(u >> 16);
}

__device__ __forceinline__ float sigmoidf_fast(float x) {
    return 1.0f / (1.0f + __expf(-x));
}

// A fragment (16x32 bf16): lane l -> row l%16; half = l/16.
// element e: K = 16*(e/8) + half*8 + (e%8) -> two contiguous 8-elem chunks.
__device__ __forceinline__ v16bf load_a_frag(const unsigned short* rowp,
                                             int kbase, int half) {
    FragBF f;
    f.u[0] = *(const us8*)(rowp + kbase + half * 8);
    f.u[1] = *(const us8*)(rowp + kbase + 16 + half * 8);
    return f.v;
}

// B fragment (32x16 bf16) from LDS: lane l -> col l%16; half = l/16.
// element e: K = 16*(e/8) + half*8 + (e%8) -> two contiguous 8-elem chunks.
__device__ __forceinline__ v16bf load_b_frag_lds(const unsigned short* ldsrow,
                                                 int half) {
    FragBF f;
    f.u[0] = *(const us8*)(ldsrow + half * 8);
    f.u[1] = *(const us8*)(ldsrow + 16 + half * 8);
    return f.v;
}

// ---------------------------------------------------------------------------
// fp32 -> bf16 bulk conversion (4 elems/thread). NT loads: f32 weights are
// read exactly once; keep them out of L2 so the bf16 copies stay resident.
// ---------------------------------------------------------------------------
__global__ __launch_bounds__(256) void cvt_f32_to_bf16(
    const float* __restrict__ src, unsigned short* __restrict__ dst, int n4) {
    int i = blockIdx.x * blockDim.x + threadIdx.x;
    if (i < n4) {
        f4 v = __builtin_nontemporal_load(((const f4*)src) + i);
        us4 o;
        o.x = f32_to_bf16(v.x);
        o.y = f32_to_bf16(v.y);
        o.z = f32_to_bf16(v.z);
        o.w = f32_to_bf16(v.w);
        ((us4*)dst)[i] = o;
    }
}

// ---------------------------------------------------------------------------
// GEMM1: encoded[b,k,l] = sigmoid(sum_f feat[b,f]*W1[k,l,f] + b1[k,l])
// grid = (B/128, K_CLS), block = 256 (8 waves). Wave w owns rows
// mblk*128 + w*16 .. +15, all 128 cols. W1[k] K-slabs double-buffered in LDS.
// ---------------------------------------------------------------------------
__global__ __launch_bounds__(256) void gemm1_enc(
    const unsigned short* __restrict__ featb,  // (B,F) bf16
    const unsigned short* __restrict__ w1b,    // (K,L,F) bf16
    const float* __restrict__ b1,              // (K,L) f32
    float* __restrict__ enc_out,               // (B,K,L) f32
    unsigned short* __restrict__ enc_b)        // (B,K,L) bf16
{
    __shared__ unsigned short ldsB[2][L_DIM * LDS_STRIDE];

    const int mblk = blockIdx.x;
    const int kcls = blockIdx.y;
    const int tid  = threadIdx.x;
    const int wave = tid >> 5;
    const int lane = tid & 31;
    const int half = lane >> 4;
    const int l16  = lane & 15;
    const int mbase = mblk * 128 + wave * 16;

    const unsigned short* arow = featb + (size_t)(mbase + l16) * F_DIM;
    const unsigned short* wk   = w1b + (size_t)kcls * L_DIM * F_DIM;

    v8f acc[8];
#pragma unroll
    for (int t = 0; t < 8; ++t)
#pragma unroll
        for (int r = 0; r < 8; ++r) acc[t][r] = 0.0f;

    // cooperative fill mapping: thread -> (row, 16-ushort chunk)
    const int fill_row = tid >> 1;           // 0..127
    const int fill_off = (tid & 1) * 16;     // 0 or 16
    const int fill_idx = fill_row * LDS_STRIDE + fill_off;
    const unsigned short* gfill = wk + (size_t)fill_row * F_DIM + fill_off;

    // prologue: stage tile 0 (direct shared indexing -> ds_store_b128)
    {
        const us8* g = (const us8*)gfill;
        *(us8*)&ldsB[0][fill_idx]     = g[0];
        *(us8*)&ldsB[0][fill_idx + 8] = g[1];
    }
    __syncthreads();

    const int NSTEP = F_DIM / 32;            // 20
    us8 pf0, pf1;
    for (int i = 0; i < NSTEP; ++i) {
        const int kk = i * 32;
        // prefetch next weight slab global -> registers (hidden by WMMAs)
        if (i + 1 < NSTEP) {
            const us8* g = (const us8*)(gfill + (i + 1) * 32);
            pf0 = g[0];
            pf1 = g[1];
        }
        v16bf a = load_a_frag(arow, kk, half);
        const unsigned short* buf = ldsB[i & 1];
#pragma unroll
        for (int g = 0; g < 2; ++g) {
            v16bf bf[4];
#pragma unroll
            for (int t = 0; t < 4; ++t)
                bf[t] = load_b_frag_lds(
                    &buf[((g * 4 + t) * 16 + l16) * LDS_STRIDE], half);
#pragma unroll
            for (int t = 0; t < 4; ++t)
                acc[g * 4 + t] = __builtin_amdgcn_wmma_f32_16x16x32_bf16(
                    false, a, false, bf[t], (short)0, acc[g * 4 + t],
                    false, false);
        }
        // stage next slab into the other buffer (safe: last read in the previous step, before its barrier)
        if (i + 1 < NSTEP) {
            *(us8*)&ldsB[(i + 1) & 1][fill_idx]     = pf0;
            *(us8*)&ldsB[(i + 1) & 1][fill_idx + 8] = pf1;
        }
        __syncthreads();
    }

    // epilogue: bias + sigmoid; f32 out is write-once -> non-temporal;
    // bf16 copy is re-read by GEMM2 -> default (L2-resident) policy.
#pragma unroll
    for (int t = 0; t < 8; ++t) {
        const int n = t * 16 + l16;
        const float bias = b1[kcls * L_DIM + n];
#pragma unroll
        for (int r = 0; r < 8; ++r) {
            const int m = mbase + half * 8 + r;
            float s = sigmoidf_fast(acc[t][r] + bias);
            const size_t idx = ((size_t)m * K_CLS + kcls) * L_DIM + n;
            __builtin_nontemporal_store(s, &enc_out[idx]);
            enc_b[idx] = f32_to_bf16(s);
        }
    }
}

// ---------------------------------------------------------------------------
// GEMM2: decoded[b,k,f] = sigmoid(sum_l enc[b,k,l]*W2[k,f,l] + b2[k,f])
// grid = (B/128, F/128, K_CLS), block = 256 (8 waves).
// ---------------------------------------------------------------------------
__global__ __launch_bounds__(256) void gemm2_dec(
    const unsigned short* __restrict__ encb,   // (B,K,L) bf16
    const unsigned short* __restrict__ w2b,    // (K,F,L) bf16
    const float* __restrict__ b2,              // (K,F) f32
    float* __restrict__ dec_out)               // (B,K,F) f32
{
    __shared__ unsigned short ldsB[2][128 * LDS_STRIDE];

    const int mblk = blockIdx.x;
    const int nblk = blockIdx.y;
    const int kcls = blockIdx.z;
    const int tid  = threadIdx.x;
    const int wave = tid >> 5;
    const int lane = tid & 31;
    const int half = lane >> 4;
    const int l16  = lane & 15;
    const int mbase = mblk * 128 + wave * 16;
    const int nbase = nblk * 128;

    const unsigned short* arow =
        encb + ((size_t)(mbase + l16) * K_CLS + kcls) * L_DIM;
    const unsigned short* wk =
        w2b + (size_t)kcls * F_DIM * L_DIM + (size_t)nbase * L_DIM;

    v8f acc[8];
#pragma unroll
    for (int t = 0; t < 8; ++t)
#pragma unroll
        for (int r = 0; r < 8; ++r) acc[t][r] = 0.0f;

    const int fill_row = tid >> 1;
    const int fill_off = (tid & 1) * 16;
    const int fill_idx = fill_row * LDS_STRIDE + fill_off;
    const unsigned short* gfill = wk + (size_t)fill_row * L_DIM + fill_off;

    {
        const us8* g = (const us8*)gfill;
        *(us8*)&ldsB[0][fill_idx]     = g[0];
        *(us8*)&ldsB[0][fill_idx + 8] = g[1];
    }
    __syncthreads();

    const int NSTEP = L_DIM / 32;            // 4
    us8 pf0, pf1;
#pragma unroll
    for (int i = 0; i < NSTEP; ++i) {
        const int kk = i * 32;
        if (i + 1 < NSTEP) {
            const us8* g = (const us8*)(gfill + (i + 1) * 32);
            pf0 = g[0];
            pf1 = g[1];
        }
        v16bf a = load_a_frag(arow, kk, half);
        const unsigned short* buf = ldsB[i & 1];
#pragma unroll
        for (int g = 0; g < 2; ++g) {
            v16bf bf[4];
#pragma unroll
            for (int t = 0; t < 4; ++t)
                bf[t] = load_b_frag_lds(
                    &buf[((g * 4 + t) * 16 + l16) * LDS_STRIDE], half);
#pragma unroll
            for (int t = 0; t < 4; ++t)
                acc[g * 4 + t] = __builtin_amdgcn_wmma_f32_16x16x32_bf16(
                    false, a, false, bf[t], (short)0, acc[g * 4 + t],
                    false, false);
        }
        if (i + 1 < NSTEP) {
            *(us8*)&ldsB[(i + 1) & 1][fill_idx]     = pf0;
            *(us8*)&ldsB[(i + 1) & 1][fill_idx + 8] = pf1;
        }
        __syncthreads();
    }

#pragma unroll
    for (int t = 0; t < 8; ++t) {
        const int f = nbase + t * 16 + l16;
        const float bias = b2[kcls * F_DIM + f];
#pragma unroll
        for (int r = 0; r < 8; ++r) {
            const int m = mbase + half * 8 + r;
            float s = sigmoidf_fast(acc[t][r] + bias);
            __builtin_nontemporal_store(
                s, &dec_out[((size_t)m * K_CLS + kcls) * F_DIM + f]);
        }
    }
}

// ---------------------------------------------------------------------------
// launch
// ---------------------------------------------------------------------------
extern "C" void kernel_launch(void* const* d_in, const int* in_sizes, int n_in,
                              void* d_out, int out_size, void* d_ws, size_t ws_size,
                              hipStream_t stream) {
    const float* feat = (const float*)d_in[0];  // (B,F)
    const float* W1   = (const float*)d_in[1];  // (K,L,F)
    const float* b1   = (const float*)d_in[2];  // (K,L)
    const float* W2   = (const float*)d_in[3];  // (K,F,L)
    const float* b2   = (const float*)d_in[4];  // (K,F)

    float* out = (float*)d_out;
    // output layout: feat (B*F) | encoded (B*K*L) | decoded (B*K*F)
    const size_t n_feat = (size_t)B_DIM * F_DIM;          //    655,360
    const size_t n_enc  = (size_t)B_DIM * K_CLS * L_DIM;  // 13,107,200
    float* enc_out = out + n_feat;
    float* dec_out = out + n_feat + n_enc;

    // workspace layout (bf16 as ushort), all offsets 256B-aligned; ~60MB total
    unsigned char* ws = (unsigned char*)d_ws;
    const size_t n_w = (size_t)K_CLS * L_DIM * F_DIM;     // 8,192,000
    unsigned short* featb = (unsigned short*)ws;
    unsigned short* w1b   = (unsigned short*)(ws + 2 * n_feat);
    unsigned short* w2b   = (unsigned short*)(ws + 2 * n_feat + 2 * n_w);
    unsigned short* encb  = (unsigned short*)(ws + 2 * n_feat + 4 * n_w);

    // feat passthrough (output[0])
    hipMemcpyAsync(out, feat, n_feat * sizeof(float),
                   hipMemcpyDeviceToDevice, stream);

    // fp32 -> bf16 conversions (element counts are multiples of 4)
    {
        int n4 = (int)(n_feat / 4);
        cvt_f32_to_bf16<<<(n4 + 255) / 256, 256, 0, stream>>>(feat, featb, n4);
    }
    {
        int n4 = (int)(n_w / 4);
        cvt_f32_to_bf16<<<(n4 + 255) / 256, 256, 0, stream>>>(W1, w1b, n4);
        cvt_f32_to_bf16<<<(n4 + 255) / 256, 256, 0, stream>>>(W2, w2b, n4);
    }

    gemm1_enc<<<dim3(B_DIM / 128, K_CLS), 256, 0, stream>>>(
        featb, w1b, b1, enc_out, encb);

    gemm2_dec<<<dim3(B_DIM / 128, F_DIM / 128, K_CLS), 256, 0, stream>>>(
        encb, w2b, b2, dec_out);
}
